// GraphUnet_46943992545660
// MI455X (gfx1250) — compile-verified, matrix operand order checked by reading, not run on hardware
//
#include <hip/hip_runtime.h>

// ---------------------------------------------------------------------------
// GraphUnet forward for MI455X (gfx1250, wave32, WMMA bf16 16x16x32,
// async global->LDS staging).  N=4096 nodes, F=320 features.
// ---------------------------------------------------------------------------

constexpr int kN  = 4096;    // nodes
constexpr int kF  = 320;     // feature dim
constexpr int kNW = kN / 32; // bitmask words per row (128)
constexpr int kLP = 40;      // LDS row pitch in halves (80B: bank-rotating, 16B-aligned)

typedef __attribute__((ext_vector_type(16))) __bf16        bf16x16;
typedef __attribute__((ext_vector_type(8)))  float         floatx8;
typedef __attribute__((ext_vector_type(4)))  unsigned int  u32x4;

union BFrag { bf16x16 v; u32x4 q[2]; };

__device__ __forceinline__ unsigned short f2bf(float x) {
  unsigned int u = __float_as_uint(x);
  unsigned int r = u + 0x7FFFu + ((u >> 16) & 1u);   // round-to-nearest-even
  return (unsigned short)(r >> 16);
}

// Async copy of 16 bytes global -> LDS (per-lane), tracked by ASYNCcnt.
__device__ __forceinline__ void async_cp_b128(unsigned short* lds_dst,
                                              const unsigned short* gsrc) {
  unsigned loff = (unsigned)(size_t)lds_dst;     // low 32 bits = LDS offset
  asm volatile("global_load_async_to_lds_b128 %0, %1, off"
               :: "v"(loff), "v"(gsrc) : "memory");
}
__device__ __forceinline__ void async_wait0() {
  asm volatile("s_wait_asynccnt 0x0" ::: "memory");
}

// Fragment load for the 16-bit WMMA 16x32 operand layout (wave32):
// lanes 0-15 take halves {0..7,16..23} of their row, lanes 16-31 take
// {8..15,24..31}: two aligned 16B LDS reads per lane.
__device__ __forceinline__ bf16x16 frag_ld(const unsigned short* rowbase, int lane) {
  BFrag f;
  const u32x4* p = (const u32x4*)(rowbase + ((lane & 16) ? 8 : 0));
  f.q[0] = p[0];
  f.q[1] = p[2];
  return f.v;
}

// ---------------------------------------------------------------------------
// Degree centrality + adjacency bitmask (one block per row, 128 threads)
// ---------------------------------------------------------------------------
__global__ void centmask_k(const float* __restrict__ A, float* __restrict__ cent,
                           unsigned int* __restrict__ maskb) {
  int row = blockIdx.x;
  int t = threadIdx.x;                      // 0..127, one 32-bit word each
  const float* ar = A + (size_t)row * kN + t * 32;
  float s = 0.f;
  unsigned int w = 0;
  #pragma unroll 4
  for (int b = 0; b < 32; ++b) {
    float v = ar[b];
    s += v;
    if (v > 0.f) w |= (1u << b);
  }
  maskb[(size_t)row * kNW + t] = w;
  __shared__ float red[128];
  red[t] = s;
  __syncthreads();
  for (int o = 64; o > 0; o >>= 1) {
    if (t < o) red[t] += red[t + o];
    __syncthreads();
  }
  if (t == 0) cent[row] = red[0] / (float)(kN - 1);
}

// ---------------------------------------------------------------------------
// Build padded bf16 H = [x0 (w0) | x1 (w1) | cent * ncent | 0-pad] (row/block)
// ---------------------------------------------------------------------------
__global__ void build_h_k(const float* __restrict__ x0, int w0,
                          const float* __restrict__ x1, int w1,
                          const float* __restrict__ cent, int ncent,
                          unsigned short* __restrict__ H, int Kpad) {
  int row = blockIdx.x;
  for (int c = threadIdx.x; c < Kpad; c += blockDim.x) {
    float v;
    if (c < w0)                   v = x0[(size_t)row * w0 + c];
    else if (c < w0 + w1)         v = x1[(size_t)row * w1 + (c - w0)];
    else if (c < w0 + w1 + ncent) v = cent[row];
    else                          v = 0.f;
    H[(size_t)row * Kpad + c] = f2bf(v);
  }
}

// Weight [Krows x 320] fp32 -> TRANSPOSED bf16 [320 x Kpad], zero-padded K.
__global__ void convWT_k(const float* __restrict__ W, int Krows, int Kpad,
                         unsigned short* __restrict__ WbT) {
  int i = blockIdx.x * blockDim.x + threadIdx.x;
  if (i >= kF * Kpad) return;
  int n = i / Kpad, k = i % Kpad;
  WbT[i] = (k < Krows) ? f2bf(W[(size_t)k * kF + n]) : (unsigned short)0;
}

// Flat fp32 -> bf16 convert (used for Uw, whose transpose-of-transpose is itself)
__global__ void tobf_k(const float* __restrict__ src,
                       unsigned short* __restrict__ dst, int n) {
  int i = blockIdx.x * blockDim.x + threadIdx.x;
  if (i < n) dst[i] = f2bf(src[i]);
}

// Wh [kN x kF] fp32 -> TRANSPOSED bf16 [kF x kN] (coalesced writes)
__global__ void tobfT_k(const float* __restrict__ src,
                        unsigned short* __restrict__ dstT) {
  int i = blockIdx.x * blockDim.x + threadIdx.x;
  if (i >= kF * kN) return;
  int c = i / kN, row = i % kN;
  dstT[i] = f2bf(src[(size_t)row * kF + c]);
}

// ---------------------------------------------------------------------------
// bf16 WMMA GEMM: C[M x 320] = A[M x K] * B[K x 320] with B given TRANSPOSED
// as BT[320 x K].  K multiple of 32.  Block = 256 threads (8 waves),
// tile 128(M) x 64(N); wave = 32x32 output.  Tiles staged via async-to-LDS.
// ---------------------------------------------------------------------------
__global__ void gemm_bf16_k(const unsigned short* __restrict__ Ab,
                            const unsigned short* __restrict__ BbT,
                            float* __restrict__ C, int K) {
  __shared__ unsigned short At[128 * kLP];   // [m][k]
  __shared__ unsigned short Bt[64 * kLP];    // [n][k]
  int tid = threadIdx.x, lane = tid & 31, wave = tid >> 5;
  int mi = wave & 3, ni = wave >> 2;         // 4 x 2 wave grid
  int Mbase = blockIdx.x * 128, Nbase = blockIdx.y * 64;

  floatx8 vzero;
  #pragma unroll
  for (int q = 0; q < 8; ++q) vzero[q] = 0.f;
  floatx8 acc[2][2];
  acc[0][0] = vzero; acc[0][1] = vzero; acc[1][0] = vzero; acc[1][1] = vzero;

  for (int kb = 0; kb < K; kb += 32) {
    // stage A tile: 128 rows x 4 chunks (16B each)
    for (int c = tid; c < 128 * 4; c += 256) {
      int rr = c >> 2, q = c & 3;
      async_cp_b128(&At[rr * kLP + q * 8],
                    Ab + (size_t)(Mbase + rr) * K + kb + q * 8);
    }
    // stage B tile: 64 rows x 4 chunks
    for (int c = tid; c < 64 * 4; c += 256) {
      int n = c >> 2, q = c & 3;
      async_cp_b128(&Bt[n * kLP + q * 8],
                    BbT + (size_t)(Nbase + n) * K + kb + q * 8);
    }
    async_wait0();
    __syncthreads();

    BFrag fb[2];
    #pragma unroll
    for (int b2 = 0; b2 < 2; ++b2)
      fb[b2].v = frag_ld(&Bt[(ni * 32 + b2 * 16 + (lane & 15)) * kLP], lane);
    #pragma unroll
    for (int a2 = 0; a2 < 2; ++a2) {
      bf16x16 fa = frag_ld(&At[(mi * 32 + a2 * 16 + (lane & 15)) * kLP], lane);
      #pragma unroll
      for (int b2 = 0; b2 < 2; ++b2)
        acc[a2][b2] = __builtin_amdgcn_wmma_f32_16x16x32_bf16(
            false, fa, false, fb[b2].v, (short)0, acc[a2][b2], false, false);
    }
    __syncthreads();
  }

  #pragma unroll
  for (int a2 = 0; a2 < 2; ++a2)
    #pragma unroll
    for (int b2 = 0; b2 < 2; ++b2) {
      int col = Nbase + ni * 32 + b2 * 16 + (lane & 15);
      #pragma unroll
      for (int v = 0; v < 8; ++v) {
        int row = Mbase + mi * 32 + a2 * 16 + v + ((lane & 16) ? 8 : 0);
        C[(size_t)row * kF + col] = acc[a2][b2][v];
      }
    }
}

// ---------------------------------------------------------------------------
// l = Wh @ a[0:320], r = Wh @ a[320:640]   (one thread per row)
// ---------------------------------------------------------------------------
__global__ void lr_k(const float* __restrict__ Wh, const float* __restrict__ a,
                     float* __restrict__ l, float* __restrict__ r) {
  int row = blockIdx.x * blockDim.x + threadIdx.x;
  if (row >= kN) return;
  const float* w = Wh + (size_t)row * kF;
  float sl = 0.f, sr = 0.f;
  for (int c = 0; c < kF; ++c) {
    float v = w[c];
    sl += v * a[c];
    sr += v * a[kF + c];
  }
  l[row] = sl;
  r[row] = sr;
}

// ---------------------------------------------------------------------------
// Pass 1: masked row softmax stats.  e_ij = lrelu(l_i + r_j), mask from
// bitmask of the ORIGINAL A via composed permutation.  8 rows / block.
// ---------------------------------------------------------------------------
__global__ void att_pass1_k(const float* __restrict__ l, const float* __restrict__ r,
                            const unsigned int* __restrict__ maskb,
                            const int* __restrict__ perm,
                            float* __restrict__ m, float* __restrict__ sinv) {
  int tid = threadIdx.x, lane = tid & 31;
  int row = blockIdx.x * 8 + (tid >> 5);
  int pi = perm[row];
  const unsigned int* mrow = maskb + (size_t)pi * kNW;
  float li = l[row];

  float mx = -3.0e38f;
  for (int j = lane; j < kN; j += 32) {
    int pj = perm[j];
    if ((mrow[pj >> 5] >> (pj & 31)) & 1u) {
      float e = li + r[j];
      e = e > 0.f ? e : 0.2f * e;
      mx = fmaxf(mx, e);
    }
  }
  #pragma unroll
  for (int o = 16; o > 0; o >>= 1) mx = fmaxf(mx, __shfl_xor(mx, o, 32));

  float s = 0.f;
  for (int j = lane; j < kN; j += 32) {
    int pj = perm[j];
    if ((mrow[pj >> 5] >> (pj & 31)) & 1u) {
      float e = li + r[j];
      e = e > 0.f ? e : 0.2f * e;
      s += __expf(e - mx);
    }
  }
  #pragma unroll
  for (int o = 16; o > 0; o >>= 1) s += __shfl_xor(s, o, 32);

  if (lane == 0) { m[row] = mx; sinv[row] = 1.0f / s; }
}

// ---------------------------------------------------------------------------
// Pass 2: out = softmax(e) @ Wh, flash-style.  Block = 256 threads = 8 waves,
// 32 output rows x 320 cols.  Wave: 16 rows x 80 cols (5 WMMA n-tiles).
// P tiles regenerated in LDS as bf16; Wh tiles async-staged from the
// transposed bf16 copy WhbT[320 x 4096] (L2-resident).
// ---------------------------------------------------------------------------
__global__ void att_pass2_k(const unsigned short* __restrict__ WhbT,
                            const float* __restrict__ l, const float* __restrict__ r,
                            const float* __restrict__ m, const float* __restrict__ sinv,
                            const unsigned int* __restrict__ maskb,
                            const int* __restrict__ perm, float* __restrict__ out) {
  __shared__ unsigned short Pl[32 * kLP];    // [row][j]  (j contiguous)
  __shared__ unsigned short Wl[kF * kLP];    // [n][j]    (j contiguous)
  __shared__ float lrow[32], mrow[32], srow[32];
  __shared__ int prow[32];

  int tid = threadIdx.x, lane = tid & 31, wave = tid >> 5;
  int mt = wave & 1;        // 16-row sub-tile
  int ng = wave >> 1;       // 80-col group
  int rowbase = blockIdx.x * 32;

  if (tid < 32) {
    int rr = rowbase + tid;
    lrow[tid] = l[rr];
    mrow[tid] = m[rr];
    srow[tid] = sinv[rr];
    prow[tid] = perm[rr];
  }

  floatx8 vzero;
  #pragma unroll
  for (int q = 0; q < 8; ++q) vzero[q] = 0.f;
  floatx8 acc[5];
  #pragma unroll
  for (int i = 0; i < 5; ++i) acc[i] = vzero;
  __syncthreads();

  for (int jb = 0; jb < kN; jb += 32) {
    // async-stage Wh^T tile: 320 rows x 4 chunks of 16B
    for (int c = tid; c < kF * 4; c += 256) {
      int n = c >> 2, q = c & 3;
      async_cp_b128(&Wl[n * kLP + q * 8],
                    WhbT + (size_t)n * kN + jb + q * 8);
    }
    // build 32x32 probability tile in bf16
    for (int e = tid; e < 32 * 32; e += 256) {
      int pr = e >> 5, pc = e & 31;
      int j = jb + pc;
      int pj = perm[j];
      unsigned int mw = maskb[(size_t)prow[pr] * kNW + (pj >> 5)];
      float pv = 0.f;
      if ((mw >> (pj & 31)) & 1u) {
        float ev = lrow[pr] + r[j];
        ev = ev > 0.f ? ev : 0.2f * ev;
        pv = __expf(ev - mrow[pr]) * srow[pr];
      }
      Pl[pr * kLP + pc] = f2bf(pv);
    }
    async_wait0();
    __syncthreads();

    bf16x16 fa = frag_ld(&Pl[(mt * 16 + (lane & 15)) * kLP], lane);
    #pragma unroll
    for (int nt = 0; nt < 5; ++nt) {
      bf16x16 fb = frag_ld(&Wl[(ng * 80 + nt * 16 + (lane & 15)) * kLP], lane);
      acc[nt] = __builtin_amdgcn_wmma_f32_16x16x32_bf16(
          false, fa, false, fb, (short)0, acc[nt], false, false);
    }
    __syncthreads();
  }

  #pragma unroll
  for (int nt = 0; nt < 5; ++nt) {
    int col = ng * 80 + nt * 16 + (lane & 15);
    #pragma unroll
    for (int v = 0; v < 8; ++v) {
      int row = rowbase + mt * 16 + v + ((lane & 16) ? 8 : 0);
      out[(size_t)row * kF + col] = acc[nt][v];
    }
  }
}

// ---------------------------------------------------------------------------
// Pooling: scores, full stable descending sort (bitonic on 64-bit keys),
// apply permutation+scale, compose permutations.
// ---------------------------------------------------------------------------
__global__ void score_k(const float* __restrict__ X, const float* __restrict__ w,
                        const float* __restrict__ b, float* __restrict__ scores) {
  int row = blockIdx.x * blockDim.x + threadIdx.x;
  if (row >= kN) return;
  const float* x = X + (size_t)row * kF;
  float s = 0.f;
  for (int c = 0; c < kF; ++c) s += x[c] * w[c];
  s = (s + b[0]) * 0.01f;
  scores[row] = 1.0f / (1.0f + __expf(-s));
}

__global__ void topk_sort_k(const float* __restrict__ scores,
                            int* __restrict__ idx, float* __restrict__ val) {
  __shared__ unsigned long long key[kN];
  int tid = threadIdx.x;  // 1024 threads
  for (int i = tid; i < kN; i += 1024) {
    unsigned int u = __float_as_uint(scores[i]);
    u = (u & 0x80000000u) ? ~u : (u | 0x80000000u);     // order-preserving map
    key[i] = ((unsigned long long)u << 32) | (unsigned int)(~i); // ties: low idx first
  }
  __syncthreads();
  for (int k2 = 2; k2 <= kN; k2 <<= 1) {
    for (int j = k2 >> 1; j > 0; j >>= 1) {
      for (int i = tid; i < kN; i += 1024) {
        int ixj = i ^ j;
        if (ixj > i) {
          bool desc = ((i & k2) == 0);
          unsigned long long a = key[i], b = key[ixj];
          if ((a < b) == desc) { key[i] = b; key[ixj] = a; }
        }
      }
      __syncthreads();
    }
  }
  for (int i = tid; i < kN; i += 1024) {
    unsigned long long kk = key[i];
    idx[i] = (int)(~(unsigned int)kk);
    unsigned int u = (unsigned int)(kk >> 32);
    u = (u & 0x80000000u) ? (u & 0x7FFFFFFFu) : ~u;
    val[i] = __uint_as_float(u);
  }
}

__global__ void pool_apply_k(const float* __restrict__ Xin, const int* __restrict__ idx,
                             const float* __restrict__ val, float* __restrict__ Xout) {
  int i = blockIdx.x * blockDim.x + threadIdx.x;
  if (i >= kN * kF) return;
  int row = i / kF, c = i % kF;
  Xout[i] = Xin[(size_t)idx[row] * kF + c] * val[row];
}

__global__ void perm_compose_k(const int* __restrict__ pin, const int* __restrict__ idx,
                               int* __restrict__ pout) {
  int i = blockIdx.x * blockDim.x + threadIdx.x;
  if (i < kN) pout[i] = pin[idx[i]];
}

__global__ void iota_k(int* __restrict__ p) {
  int i = blockIdx.x * blockDim.x + threadIdx.x;
  if (i < kN) p[i] = i;
}

// Unpool scatter: out[idx[r]] = X[r] * sigmoid(G[r] + b)
__global__ void unpool_scatter_k(const float* __restrict__ X, const float* __restrict__ G,
                                 const float* __restrict__ b, const int* __restrict__ idx,
                                 float* __restrict__ out) {
  int i = blockIdx.x * blockDim.x + threadIdx.x;
  if (i >= kN * kF) return;
  int row = i / kF, c = i % kF;
  float aw = 1.0f / (1.0f + __expf(-(G[i] + b[c])));
  out[(size_t)idx[row] * kF + c] = X[i] * aw;
}

__global__ void zero_k(float* __restrict__ p, int n) {
  int i = blockIdx.x * blockDim.x + threadIdx.x;
  if (i < n) p[i] = 0.f;
}
__global__ void elu_k(float* __restrict__ p, int n) {
  int i = blockIdx.x * blockDim.x + threadIdx.x;
  if (i < n) { float x = p[i]; p[i] = x > 0.f ? x : (__expf(x) - 1.0f); }
}
__global__ void add_k(float* __restrict__ d, const float* __restrict__ s, int n) {
  int i = blockIdx.x * blockDim.x + threadIdx.x;
  if (i < n) d[i] += s[i];
}

// ---------------------------------------------------------------------------
// Host orchestration
// ---------------------------------------------------------------------------
extern "C" void kernel_launch(void* const* d_in, const int* in_sizes, int n_in,
                              void* d_out, int out_size, void* d_ws, size_t ws_size,
                              hipStream_t stream) {
  (void)in_sizes; (void)n_in; (void)out_size; (void)ws_size;
  const float* A        = (const float*)d_in[0];
  const float* X0       = (const float*)d_in[1];
  const float* start_W  = (const float*)d_in[2];
  const float* start_a  = (const float*)d_in[3];
  const float* bottom_W = (const float*)d_in[4];
  const float* bottom_a = (const float*)d_in[5];
  const float* end_W    = (const float*)d_in[6];
  const float* end_a    = (const float*)d_in[7];
  const float* down_W[2]   = {(const float*)d_in[8],  (const float*)d_in[16]};
  const float* down_a[2]   = {(const float*)d_in[9],  (const float*)d_in[17]};
  const float* up_W[2]     = {(const float*)d_in[10], (const float*)d_in[18]};
  const float* up_a[2]     = {(const float*)d_in[11], (const float*)d_in[19]};
  const float* pool_w[2]   = {(const float*)d_in[12], (const float*)d_in[20]};
  const float* pool_b[2]   = {(const float*)d_in[13], (const float*)d_in[21]};
  const float* unpool_w[2] = {(const float*)d_in[14], (const float*)d_in[22]};
  const float* unpool_b[2] = {(const float*)d_in[15], (const float*)d_in[23]};

  // workspace carve
  char* wp = (char*)d_ws;
  auto carve = [&](size_t bytes) -> void* {
    void* r = (void*)wp;
    wp += (bytes + 255) & ~(size_t)255;
    return r;
  };
  float*          cent  = (float*)carve((size_t)kN * 4);
  unsigned int*   maskb = (unsigned int*)carve((size_t)kN * kNW * 4);
  unsigned short* Hb    = (unsigned short*)carve((size_t)kN * 672 * 2);
  unsigned short* WbT   = (unsigned short*)carve((size_t)kF * 672 * 2);
  float*          Wh    = (float*)carve((size_t)kN * kF * 4);
  unsigned short* WhbT  = (unsigned short*)carve((size_t)kF * kN * 2);
  float* lb   = (float*)carve(kN * 4);
  float* rb   = (float*)carve(kN * 4);
  float* mb   = (float*)carve(kN * 4);
  float* sb   = (float*)carve(kN * 4);
  float* XA   = (float*)carve((size_t)kN * kF * 4);
  float* XB   = (float*)carve((size_t)kN * kF * 4);
  float* dn0  = (float*)carve((size_t)kN * kF * 4);
  float* dn1  = (float*)carve((size_t)kN * kF * 4);
  float* Gt   = (float*)carve((size_t)kN * kF * 4);
  float* sc   = (float*)carve(kN * 4);
  int* perm0  = (int*)carve(kN * 4);
  int* perm1  = (int*)carve(kN * 4);
  int* perm2  = (int*)carve(kN * 4);
  int* idx0   = (int*)carve(kN * 4);
  int* idx1   = (int*)carve(kN * 4);
  float* val0 = (float*)carve(kN * 4);
  float* val1 = (float*)carve(kN * 4);

  const int NF = kN * kF;
  dim3 b256(256);

  // preprocessing: centrality + adjacency bitmask + identity permutation
  centmask_k<<<dim3(kN), dim3(128), 0, stream>>>(A, cent, maskb);
  iota_k<<<dim3(kN / 256), b256, 0, stream>>>(perm0);

  auto run_gat = [&](const float* x0, int w0, const float* x1, int w1, int ncent,
                     const float* Wmat, int Krows, int Kpad, const float* avec,
                     const int* perm, float* outp, bool do_elu) {
    build_h_k<<<dim3(kN), b256, 0, stream>>>(x0, w0, x1, w1, cent, ncent, Hb, Kpad);
    convWT_k<<<dim3((kF * Kpad) / 256), b256, 0, stream>>>(Wmat, Krows, Kpad, WbT);
    gemm_bf16_k<<<dim3(kN / 128, kF / 64), b256, 0, stream>>>(Hb, WbT, Wh, Kpad);
    lr_k<<<dim3(kN / 256), b256, 0, stream>>>(Wh, avec, lb, rb);
    tobfT_k<<<dim3(NF / 256), b256, 0, stream>>>(Wh, WhbT);
    att_pass1_k<<<dim3(kN / 8), b256, 0, stream>>>(lb, rb, maskb, perm, mb, sb);
    att_pass2_k<<<dim3(kN / 32), b256, 0, stream>>>(WhbT, lb, rb, mb, sb, maskb, perm, outp);
    if (do_elu) elu_k<<<dim3(NF / 256), b256, 0, stream>>>(outp, NF);
  };

  auto run_pool = [&](const float* Xin, float* Xout, const float* pw, const float* pb,
                      int* idx, float* val, const int* pin, int* pout) {
    score_k<<<dim3(kN / 256), b256, 0, stream>>>(Xin, pw, pb, sc);
    topk_sort_k<<<dim3(1), dim3(1024), 0, stream>>>(sc, idx, val);
    pool_apply_k<<<dim3(NF / 256), b256, 0, stream>>>(Xin, idx, val, Xout);
    perm_compose_k<<<dim3(kN / 256), b256, 0, stream>>>(pin, idx, pout);
  };

  auto run_unpool = [&](const float* Xin, float* Xout, const float* Uw,
                        const float* Ub, const int* idx) {
    build_h_k<<<dim3(kN), b256, 0, stream>>>(Xin, kF, (const float*)nullptr, 0,
                                             cent, 0, Hb, kF);         // bf16 convert
    // B = Uw^T, so B^T = Uw: plain convert gives the transposed-B layout
    tobf_k<<<dim3((kF * kF) / 256), b256, 0, stream>>>(Uw, WbT, kF * kF);
    gemm_bf16_k<<<dim3(kN / 128, kF / 64), b256, 0, stream>>>(Hb, WbT, Gt, kF);
    zero_k<<<dim3(NF / 256), b256, 0, stream>>>(Xout, NF);
    unpool_scatter_k<<<dim3(NF / 256), b256, 0, stream>>>(Xin, Gt, Ub, idx, Xout);
  };

  float* startout = (float*)d_out + (size_t)kN * kF;  // second return value

  // start GAT: h = [X0 | cent | cent] (K=322)
  run_gat(X0, kF, nullptr, 0, 2, start_W, 322, 352, start_a, perm0, startout, true);

  // down level 0
  run_gat(startout, kF, nullptr, 0, 1, down_W[0], 321, 352, down_a[0], perm0, dn0, true);
  run_pool(dn0, XA, pool_w[0], pool_b[0], idx0, val0, perm0, perm1);

  // down level 1
  run_gat(XA, kF, nullptr, 0, 1, down_W[1], 321, 352, down_a[1], perm1, dn1, true);
  run_pool(dn1, XB, pool_w[1], pool_b[1], idx1, val1, perm1, perm2);

  // bottom
  run_gat(XB, kF, nullptr, 0, 1, bottom_W, 321, 352, bottom_a, perm2, XA, true);

  // up 0 (level 2 -> 1)
  run_unpool(XA, XB, unpool_w[0], unpool_b[0], idx1);
  run_gat(XB, kF, nullptr, 0, 1, up_W[0], 321, 352, up_a[0], perm1, XA, true);
  add_k<<<dim3(NF / 256), b256, 0, stream>>>(XA, dn1, NF);

  // up 1 (level 1 -> 0)
  run_unpool(XA, XB, unpool_w[1], unpool_b[1], idx0);
  run_gat(XB, kF, nullptr, 0, 1, up_W[1], 321, 352, up_a[1], perm0, XA, true);
  add_k<<<dim3(NF / 256), b256, 0, stream>>>(XA, dn0, NF);

  // end GAT: h = [X | org_X | cent] (K=641), no elu, writes first output
  run_gat(XA, kF, startout, kF, 1, end_W, 641, 672, end_a, perm0, (float*)d_out, false);
}
